// TransGNN_3341484556840
// MI455X (gfx1250) — compile-verified, matrix operand order checked by reading, not run on hardware
//
#include <hip/hip_runtime.h>

typedef __attribute__((ext_vector_type(16))) _Float16 v16h;
typedef __attribute__((ext_vector_type(8)))  _Float16 v8h;
typedef __attribute__((ext_vector_type(8)))  float    v8f;

// ---------------- wave helpers ----------------
__device__ __forceinline__ float warp_sum(float x) {
#pragma unroll
  for (int o = 16; o > 0; o >>= 1) x += __shfl_xor(x, o, 32);
  return x;
}
__device__ __forceinline__ float warp_max(float x) {
#pragma unroll
  for (int o = 16; o > 0; o >>= 1) x = fmaxf(x, __shfl_xor(x, o, 32));
  return x;
}

// Fragment layout (ISA 05_wmma 16-bit A 16x32): lane<16 -> K in [0,8)+[16,24),
// lane>=16 -> K in [8,16)+[24,32); row (A) / col (B^T) = lane&15.
__device__ __forceinline__ v16h frag_f16(const _Float16* p) {
  v8h lo = *(const v8h*)(p);
  v8h hi = *(const v8h*)(p + 16);
  return __builtin_shufflevector(lo, hi, 0, 1, 2, 3, 4, 5, 6, 7,
                                 8, 9, 10, 11, 12, 13, 14, 15);
}

// ---------------- WMMA GEMM with fused epilogues ----------------
// One wave computes a 16 x (16*NB) strip; f32 accumulate.
// EP: 0=+bias 1=proj blend 2=score mix 3=simnorm accum 4=sigmoid gate
//     5=relu+bias 6=bias+residual 7=scale-store
// AM: 0 = A f32 row-major [M,K];  1 = A f16 row-major [M,K]
// BM: 2 = B f16 fragment-swizzled; 3 = B f16 transposed row-major [N,K]
template <int EP, int AM, int BM, int NB>
__global__ void __launch_bounds__(32)
gemm16(const float* __restrict__ A, int lda,
       const float* __restrict__ B, int ldb,
       float* __restrict__ O, int ldo, int K,
       const float* __restrict__ bias,
       const float* __restrict__ ex0,
       const float* __restrict__ ex1,
       const float* __restrict__ ascale,
       const float* __restrict__ alphap,
       int ip0, float fscale)
{
  const int lane = threadIdx.x;
  const int hi   = lane >> 4;
  const int l15  = lane & 15;
  const int koff = hi * 8;
  const int tM   = blockIdx.y * 16;
  const int tN0  = blockIdx.x * 16 * NB;
  const int ktiles = K >> 5;

  v8f c[NB];
#pragma unroll
  for (int n = 0; n < NB; ++n) c[n] = (v8f){0.f,0.f,0.f,0.f,0.f,0.f,0.f,0.f};

  for (int k0 = 0; k0 < K; k0 += 32) {
    v16h a;
    if (AM == 0) {
      const float* ap = A + (size_t)(tM + l15) * lda + k0 + koff;
#pragma unroll
      for (int e = 0; e < 8; ++e) {
        float f0 = ap[e], f1 = ap[16 + e];
        if (EP == 3) { f0 *= ascale[k0 + koff + e]; f1 *= ascale[k0 + koff + 16 + e]; }
        a[e] = (_Float16)f0; a[e + 8] = (_Float16)f1;
      }
    } else {
      a = frag_f16((const _Float16*)A + (size_t)(tM + l15) * lda + k0 + koff);
    }
#pragma unroll
    for (int n = 0; n < NB; ++n) {
      const int tN = tN0 + n * 16;
      v16h b;
      if (BM == 3) {
        b = frag_f16((const _Float16*)B + (size_t)(tN + l15) * ldb + k0 + koff);
      } else {  // BM==2: one 32-byte per-lane load, 1KB coalesced per tile
        b = *(const v16h*)((const _Float16*)B +
              (((size_t)(tN >> 4) * ktiles) + (k0 >> 5)) * 512 + lane * 16);
      }
      c[n] = __builtin_amdgcn_wmma_f32_16x16x32_f16(false, a, false, b, (short)0,
                                                    c[n], false, false);
    }
  }

#pragma unroll
  for (int n = 0; n < NB; ++n) {
    const int tN = tN0 + n * 16;
#pragma unroll
    for (int v = 0; v < 8; ++v) {
      const int r  = tM + v + hi * 8;
      const int cc = tN + l15;
      const float acc = c[n][v];
      const size_t oidx = (size_t)r * ldo + cc;
      if (EP == 0) {
        O[oidx] = acc + bias[cc];
      } else if (EP == 1) {
        const float al = *alphap;
        float lv = (r < ip0) ? ex0[(size_t)r * 128 + cc]
                             : ex1[(size_t)(r - ip0) * 128 + cc];
        O[oidx] = al * (acc + bias[cc]) + (1.f - al) * lv;
      } else if (EP == 2) {
        O[oidx] = 0.5f * acc + 0.5f * ex0[oidx];
      } else if (EP == 3) {
        O[oidx] += (*alphap) * ex1[r] * acc;
      } else if (EP == 4) {
        float g = 1.f / (1.f + __expf(-(acc + bias[cc])));
        O[oidx] = ex0[oidx] + g * ex1[oidx];
      } else if (EP == 5) {
        float t = acc + bias[cc];
        O[oidx] = t > 0.f ? t : 0.f;
      } else if (EP == 6) {
        O[oidx] = acc + bias[cc] + ex0[oidx];
      } else {
        O[oidx] = fscale * acc;
      }
    }
  }
}

// ---------------- B-operand fragment swizzler: f32 [K,C] -> f16 tiles --------
__global__ void __launch_bounds__(32)
swz_b_kernel(const float* __restrict__ B, int ldb, int K, _Float16* __restrict__ out)
{
  const int ct = blockIdx.x, ks = blockIdx.y, lane = threadIdx.x;
  const int hi = lane >> 4, l15 = lane & 15, koff = hi * 8;
  const int ktiles = K >> 5;
  v16h b;
#pragma unroll
  for (int e = 0; e < 16; ++e) {
    int kk = ks * 32 + koff + ((e < 8) ? e : e + 8);
    b[e] = (_Float16)B[(size_t)kk * ldb + ct * 16 + l15];
  }
  *(v16h*)(out + ((size_t)ct * ktiles + ks) * 512 + lane * 16) = b;
}

// ---------------- plain f32 -> f16 row-major copy ----------------
__global__ void cvt16_kernel(const float* __restrict__ in, _Float16* __restrict__ out, int n)
{
  int i = blockIdx.x * blockDim.x + threadIdx.x;
  if (i < n) out[i] = (_Float16)in[i];
}

// ---------------- positional encoding table ----------------
__global__ void pe_kernel(float* __restrict__ pe, int n, int d) {
  int i = blockIdx.x * blockDim.x + threadIdx.x;
  if (i >= n * d) return;
  int row = i / d, col = i % d;
  float div = __expf((float)(col & ~1) * (-9.210340371976184f / (float)d));
  float ang = (float)row * div;
  pe[i] = (col & 1) ? cosf(ang) : sinf(ang);
}

// ---------------- degree^-1/2 of visual_sim ----------------
__global__ void __launch_bounds__(32)
dinv_kernel(const float* __restrict__ vs, float* __restrict__ dinv, int I) {
  int r = blockIdx.x, lane = threadIdx.x;
  const float* row = vs + (size_t)r * I;
  float s = 0.f;
  for (int j = lane; j < I; j += 32) s += row[j];
  s = warp_sum(s);
  if (lane == 0) dinv[r] = rsqrtf(s + 1e-7f);
}

// ---------------- streaming top-K + sparse gather ----------------
__global__ void __launch_bounds__(32)
topk_gather_kernel(const float* __restrict__ score, const float* __restrict__ adj,
                   const float* __restrict__ prev, float* __restrict__ cur,
                   int rowbase, int Ntot)
{
  const int r = blockIdx.x, lane = threadIdx.x;
  const float* srow = score + (size_t)r * Ntot;
  float val[10]; int idx[10];
#pragma unroll
  for (int t = 0; t < 10; ++t) { val[t] = -__builtin_inff(); idx[t] = 0x7fffffff; }
  for (int j = lane; j < Ntot; j += 32) {
    float v = srow[j];
    if (v > val[9]) {
      val[9] = v; idx[9] = j;
#pragma unroll
      for (int t = 9; t > 0; --t) {
        if (val[t] > val[t - 1]) {
          float tv = val[t]; val[t] = val[t - 1]; val[t - 1] = tv;
          int ti = idx[t]; idx[t] = idx[t - 1]; idx[t - 1] = ti;
        }
      }
    }
  }
  int p = 0, chosen[10];
#pragma unroll
  for (int t = 0; t < 10; ++t) {
    float bv = (p < 10) ? val[p] : -__builtin_inff();
    int   bi = (p < 10) ? idx[p] : 0x7fffffff;
#pragma unroll
    for (int o = 16; o > 0; o >>= 1) {
      float ov = __shfl_xor(bv, o, 32);
      int   oi = __shfl_xor(bi, o, 32);
      if (ov > bv || (ov == bv && oi < bi)) { bv = ov; bi = oi; }
    }
    chosen[t] = bi;
    if (p < 10 && idx[p] == bi) ++p;
  }
  const int grow = rowbase + r;
  const float* arow = adj + (size_t)grow * Ntot;
  float4 acc = make_float4(0.f, 0.f, 0.f, 0.f);
#pragma unroll
  for (int t = 0; t < 10; ++t) {
    int j = chosen[t];
    float a = arow[j];
    if (a != 0.f) {
      float4 pv = ((const float4*)(prev + (size_t)j * 128))[lane];
      acc.x += a * pv.x; acc.y += a * pv.y; acc.z += a * pv.z; acc.w += a * pv.w;
    }
  }
  ((float4*)(cur + (size_t)grow * 128))[lane] = acc;
}

// -------- row softmax over S columns; emits f16 probabilities ---------------
__global__ void __launch_bounds__(32)
softmax_rows(const float* __restrict__ att, _Float16* __restrict__ att16, int S)
{
  const int r = blockIdx.x, lane = threadIdx.x;
  const float* row = att + (size_t)r * S;
  _Float16* orow = att16 + (size_t)r * S;
  float mx = -__builtin_inff();
  for (int j = lane; j < S; j += 32) mx = fmaxf(mx, row[j]);
  mx = warp_max(mx);
  float sum = 0.f;
  for (int j = lane; j < S; j += 32) sum += __expf(row[j] - mx);
  sum = warp_sum(sum);
  const float inv = 1.f / sum;
  for (int j = lane; j < S; j += 32)
    orow[j] = (_Float16)(__expf(row[j] - mx) * inv);
}

// ---------------- post-LN layernorm over D=128 ----------------
__global__ void __launch_bounds__(32)
layernorm_kernel(const float* __restrict__ X, float* __restrict__ O,
                 const float* __restrict__ g, const float* __restrict__ b)
{
  const int r = blockIdx.x, lane = threadIdx.x;
  float4 x = ((const float4*)(X + (size_t)r * 128))[lane];
  float s  = x.x + x.y + x.z + x.w;
  float ss = x.x * x.x + x.y * x.y + x.z * x.z + x.w * x.w;
  s = warp_sum(s); ss = warp_sum(ss);
  float mean = s * (1.f / 128.f);
  float var  = ss * (1.f / 128.f) - mean * mean;
  float rinv = rsqrtf(var + 1e-5f);
  float4 o;
  o.x = (x.x - mean) * rinv * g[lane * 4 + 0] + b[lane * 4 + 0];
  o.y = (x.y - mean) * rinv * g[lane * 4 + 1] + b[lane * 4 + 1];
  o.z = (x.z - mean) * rinv * g[lane * 4 + 2] + b[lane * 4 + 2];
  o.w = (x.w - mean) * rinv * g[lane * 4 + 3] + b[lane * 4 + 3];
  ((float4*)(O + (size_t)r * 128))[lane] = o;
}

// ---------------- final = e0 + 0.75*e1 + 0.5*e2; out = [final|final] --------
__global__ void final_combine(const float* __restrict__ e0, const float* __restrict__ e1,
                              const float* __restrict__ e2, float* __restrict__ out, int ND)
{
  int i = blockIdx.x * blockDim.x + threadIdx.x;
  if (i >= ND) return;
  float f = e0[i] + 0.75f * e1[i] + 0.5f * e2[i];
  out[i] = f;
  out[ND + i] = f;
}

// ---------------- host orchestration ----------------
extern "C" void kernel_launch(void* const* d_in, const int* in_sizes, int n_in,
                              void* d_out, int out_size, void* d_ws, size_t ws_size,
                              hipStream_t stream)
{
  const int N = 6144, U = 3072, I = 3072, D = 128, RAW = 1280, DFF = 512, CH = 512;
  const float RSQRT_HD = 0.17677669529663687f;
  const float* adj      = (const float*)d_in[0];
  const float* raw      = (const float*)d_in[1];
  const float* proj_w   = (const float*)d_in[2];
  const float* proj_b   = (const float*)d_in[3];
  const float* user_emb = (const float*)d_in[4];
  const float* item_emb = (const float*)d_in[5];
  const float* vsim     = (const float*)d_in[6];
  const float* alphap   = (const float*)d_in[7];
  const float* gate_w   = (const float*)d_in[8];
  const float* gate_b   = (const float*)d_in[9];
  const float* EW[2][8];
  for (int s = 0; s < 2; ++s)
    for (int j = 0; j < 8; ++j) EW[s][j] = (const float*)d_in[10 + 8 * s + j];

  // ---- f32 workspace ----
  float* W = (float*)d_ws;
  size_t off = 0;
  auto alloc = [&](size_t n) { float* p = W + off; off += n; return p; };
  const size_t ND = (size_t)N * D, SD = (size_t)U * D;
  float* pe   = alloc(ND);
  float* e0   = alloc(ND);
  float* e1   = alloc(ND);
  float* e2   = alloc(ND);
  float* cur  = alloc(ND);
  float* curg = alloc(ND);
  float* qb   = alloc(SD);
  float* kb   = alloc(SD);
  float* vb   = alloc(SD);
  float* o1   = alloc(SD);
  float* tb   = alloc(SD);
  float* x1   = alloc(SD);
  float* hb   = alloc((size_t)U * DFF);
  float* scr  = alloc((size_t)CH * N);
  float* att  = alloc((size_t)U * U);            // f32 scores (L2-resident)
  float* dinv = alloc(I);
  float* ebuf[3] = { e0, e1, e2 };

  // ---- f16 workspace (4KB-aligned tail of the f32 region) ----
  _Float16* HB = (_Float16*)(W + off);
  size_t hoff = 0;
  auto halloc = [&](size_t n) { _Float16* p = HB + hoff; hoff += n; return p; };
  _Float16* w_proj_s = halloc((size_t)RAW * D);      // swizzled weights
  _Float16* w_gate_s = halloc((size_t)D * D);
  _Float16* w_aw_s[2]  = { halloc((size_t)4 * D * D), halloc((size_t)4 * D * D) };
  _Float16* w_w1_s[2]  = { halloc((size_t)D * DFF),   halloc((size_t)D * DFF) };
  _Float16* w_w2_s[2]  = { halloc((size_t)DFF * D),   halloc((size_t)DFF * D) };
  _Float16* b_sim_s  = halloc(SD);                   // swizzled prev[U:]
  _Float16* b_v_s    = halloc(SD);                   // swizzled v (per encoder)
  _Float16* prev16   = halloc(ND);                   // row-major f16 copies
  _Float16* q16      = halloc(SD);
  _Float16* k16      = halloc(SD);
  _Float16* att16    = halloc((size_t)U * U);        // f16 probabilities

  const dim3 w32(32);
  pe_kernel<<<(int)((ND + 255) / 256), 256, 0, stream>>>(pe, N, D);
  dinv_kernel<<<I, w32, 0, stream>>>(vsim, dinv, I);
  // one-time weight swizzles
  swz_b_kernel<<<dim3(D / 16, RAW / 32), w32, 0, stream>>>(proj_w, D, RAW, w_proj_s);
  swz_b_kernel<<<dim3(D / 16, D / 32), w32, 0, stream>>>(gate_w, D, D, w_gate_s);
  for (int s = 0; s < 2; ++s) {
    for (int m = 0; m < 4; ++m)
      swz_b_kernel<<<dim3(D / 16, D / 32), w32, 0, stream>>>(
          EW[s][0] + m * D * D, D, D, w_aw_s[s] + (size_t)m * D * D);
    swz_b_kernel<<<dim3(DFF / 16, D / 32), w32, 0, stream>>>(EW[s][2], DFF, D, w_w1_s[s]);
    swz_b_kernel<<<dim3(D / 16, DFF / 32), w32, 0, stream>>>(EW[s][4], D, DFF, w_w2_s[s]);
  }
  // x = alpha*(raw@proj_w + b) + (1-alpha)*concat(user_emb,item_emb)
  gemm16<1, 0, 2, 4><<<dim3(D / 64, N / 16), w32, 0, stream>>>(
      raw, RAW, (const float*)w_proj_s, D, e0, D, RAW,
      proj_b, user_emb, item_emb, nullptr, alphap, U, 0.f);

  for (int blk = 0; blk < 2; ++blk) {
    const float* prev = ebuf[blk];
    float* nxt = ebuf[blk + 1];
    cvt16_kernel<<<(int)((ND + 255) / 256), 256, 0, stream>>>(prev, prev16, (int)ND);
    // top-K sampled message passing: f16 e@e.T (chunked) + streaming top-10
    for (int cb = 0; cb < N / CH; ++cb) {
      const int rb = cb * CH;
      gemm16<2, 1, 3, 4><<<dim3(N / 64, CH / 16), w32, 0, stream>>>(
          (const float*)(prev16 + (size_t)rb * D), D, (const float*)prev16, D,
          scr, N, D, nullptr, adj + (size_t)rb * N, nullptr, nullptr, alphap, 0, 0.f);
      topk_gather_kernel<<<CH, w32, 0, stream>>>(scr, adj, prev, cur, rb, N);
    }
    // cur[U:] += alpha * dinv[i]*vs[i,j]*dinv[j] @ prev[U:]   (sim_norm fused)
    swz_b_kernel<<<dim3(D / 16, U / 32), w32, 0, stream>>>(prev + (size_t)U * D, D, U, b_sim_s);
    gemm16<3, 0, 2, 4><<<dim3(D / 64, I / 16), w32, 0, stream>>>(
        vsim, I, (const float*)b_sim_s, D, cur + (size_t)U * D, D, I,
        nullptr, nullptr, dinv, dinv, alphap, 0, 0.f);
    // gated positional encoding
    gemm16<4, 0, 2, 4><<<dim3(D / 64, N / 16), w32, 0, stream>>>(
        cur, D, (const float*)w_gate_s, D, curg, D, D,
        gate_b, cur, pe, nullptr, alphap, 0, 0.f);
    // two transformer encoders (users rows [0,U), items rows [U,N))
    for (int s = 0; s < 2; ++s) {
      const float* ab = EW[s][1]; const float* b1 = EW[s][3];
      const float* b2 = EW[s][5]; const float* lg = EW[s][6]; const float* lb = EW[s][7];
      const _Float16* aws = w_aw_s[s];
      const float* xin = curg + (size_t)s * U * D;
      gemm16<0, 0, 2, 4><<<dim3(D / 64, U / 16), w32, 0, stream>>>(
          xin, D, (const float*)(aws + 0 * D * D), D, qb, D, D,
          ab + 0 * D, nullptr, nullptr, nullptr, alphap, 0, 0.f);
      gemm16<0, 0, 2, 4><<<dim3(D / 64, U / 16), w32, 0, stream>>>(
          xin, D, (const float*)(aws + 1 * D * D), D, kb, D, D,
          ab + 1 * D, nullptr, nullptr, nullptr, alphap, 0, 0.f);
      gemm16<0, 0, 2, 4><<<dim3(D / 64, U / 16), w32, 0, stream>>>(
          xin, D, (const float*)(aws + 2 * D * D), D, vb, D, D,
          ab + 2 * D, nullptr, nullptr, nullptr, alphap, 0, 0.f);
      cvt16_kernel<<<(int)((SD + 255) / 256), 256, 0, stream>>>(qb, q16, (int)SD);
      cvt16_kernel<<<(int)((SD + 255) / 256), 256, 0, stream>>>(kb, k16, (int)SD);
      swz_b_kernel<<<dim3(D / 16, U / 32), w32, 0, stream>>>(vb, D, U, b_v_s);
      // attention: WMMA q@k^T -> row softmax (f16 out) -> WMMA attn@V,
      // score matrices stay resident in the 192MB L2 across the three passes.
      for (int h = 0; h < 4; ++h) {
        gemm16<7, 1, 3, 4><<<dim3(U / 64, U / 16), w32, 0, stream>>>(
            (const float*)(q16 + h * 32), D, (const float*)(k16 + h * 32), D,
            att, U, 32, nullptr, nullptr, nullptr, nullptr, alphap, 0, RSQRT_HD);
        softmax_rows<<<U, w32, 0, stream>>>(att, att16, U);
        gemm16<7, 1, 2, 2><<<dim3(1, U / 16), w32, 0, stream>>>(
            (const float*)att16, U, (const float*)(b_v_s + (size_t)(h * 2) * (U / 32) * 512),
            D, o1 + h * 32, D, U, nullptr, nullptr, nullptr, nullptr, alphap, 0, 1.f);
      }
      gemm16<6, 0, 2, 4><<<dim3(D / 64, U / 16), w32, 0, stream>>>(     // x + O-proj
          o1, D, (const float*)(aws + 3 * D * D), D, tb, D, D,
          ab + 3 * D, xin, nullptr, nullptr, alphap, 0, 0.f);
      layernorm_kernel<<<U, w32, 0, stream>>>(tb, x1, lg, lb);
      gemm16<5, 0, 2, 4><<<dim3(DFF / 64, U / 16), w32, 0, stream>>>(   // relu FFN1
          x1, D, (const float*)w_w1_s[s], DFF, hb, DFF, D,
          b1, nullptr, nullptr, nullptr, alphap, 0, 0.f);
      gemm16<6, 0, 2, 4><<<dim3(D / 64, U / 16), w32, 0, stream>>>(     // x1 + FFN2
          hb, DFF, (const float*)w_w2_s[s], D, tb, D, DFF,
          b2, x1, nullptr, nullptr, alphap, 0, 0.f);
      layernorm_kernel<<<U, w32, 0, stream>>>(tb, nxt + (size_t)s * U * D, lg + D, lb + D);
    }
  }
  final_combine<<<(int)((ND + 255) / 256), 256, 0, stream>>>(e0, e1, e2, (float*)d_out, (int)ND);
}
